// SwinTransformerBlock_47175920780034
// MI455X (gfx1250) — matrix operand order repaired and checked by reading, not compiled, not run on hardware
//
#include <hip/hip_runtime.h>

typedef __attribute__((ext_vector_type(16))) __bf16 bf16x16;
typedef __attribute__((ext_vector_type(8)))  float   floatx8;

#define LN_EPS 1e-5f

// Problem dims (fixed by reference)
constexpr int NWIN      = 128;     // B * (32/8)^3
constexpr int NELEM     = 512;     // 8^3 tokens per window
constexpr int CDIM      = 96;
constexpr int HDIM      = 32;
constexpr int ROWS      = 65536;   // B * 32^3
constexpr int HIDDEN    = 384;
constexpr int TBLN      = 3375;    // 15^3

// ---------- WMMA fragment helpers (contiguous k = 16*hl + e in both A and B) ----------
__device__ __forceinline__ int kpat(int e, int hl) { return 16 * hl + e; }

__device__ __forceinline__ bf16x16 a_from_f32(const float* __restrict__ p, int hl) {
  bf16x16 a;
#pragma unroll
  for (int e = 0; e < 16; ++e) a[e] = (__bf16)p[kpat(e, hl)];
  return a;
}
__device__ __forceinline__ bf16x16 a_from_bf16(const __bf16* __restrict__ p, int hl) {
  bf16x16 a;
#pragma unroll
  for (int e = 0; e < 16; ++e) a[e] = p[kpat(e, hl)];
  return a;
}
__device__ __forceinline__ bf16x16 b_from_f32(const float* __restrict__ p) {
  bf16x16 b;
#pragma unroll
  for (int e = 0; e < 16; ++e) b[e] = (__bf16)p[e];
  return b;
}
__device__ __forceinline__ bf16x16 b_from_bf16(const __bf16* __restrict__ p) {
  bf16x16 b;
#pragma unroll
  for (int e = 0; e < 16; ++e) b[e] = p[e];
  return b;
}
__device__ __forceinline__ floatx8 wmma_bf16(bf16x16 a, bf16x16 b, floatx8 c) {
  return __builtin_amdgcn_wmma_f32_16x16x32_bf16(false, a, false, b, (short)0, c, false, false);
}

__device__ __forceinline__ float wave_sum(float v) {
#pragma unroll
  for (int o = 16; o > 0; o >>= 1) v += __shfl_xor(v, o, 32);
  return v;
}

// sign-preserving log-CPB transform: sign(v)*log2(|v|+1)/log2(8)
__device__ __forceinline__ float cpb_xform(float v) {
  return copysignf(log2f(fabsf(v) + 1.0f) * (1.0f / 3.0f), v);
}

// ---------- Kernel 1: CPB MLP -> raw bias table (3375 x 3) ----------
__global__ void cpb_kernel(const float* __restrict__ w1, const float* __restrict__ b1,
                           const float* __restrict__ w2, float* __restrict__ tbl) {
  int t = blockIdx.x * blockDim.x + threadIdx.x;
  if (t >= TBLN) return;
  int i = t / 225, j = (t / 15) % 15, k = t % 15;
  float c0 = cpb_xform((float)(i - 7) * (8.0f / 7.0f));
  float c1 = cpb_xform((float)(j - 7) * (8.0f / 7.0f));
  float c2 = cpb_xform((float)(k - 7) * 8.0f);   // channel 2 not normalized (faithful)
  float a0 = 0.f, a1 = 0.f, a2 = 0.f;
  for (int h = 0; h < 512; ++h) {
    float hv = fmaxf(w1[h * 3] * c0 + w1[h * 3 + 1] * c1 + w1[h * 3 + 2] * c2 + b1[h], 0.f);
    a0 += w2[h] * hv;
    a1 += w2[512 + h] * hv;
    a2 += w2[1024 + h] * hv;
  }
  tbl[t * 3 + 0] = a0; tbl[t * 3 + 1] = a1; tbl[t * 3 + 2] = a2;
}

// ---------- Kernel 2: QKV GEMM (WMMA), roll+window gather fused ----------
__global__ __launch_bounds__(256)
void qkv_kernel(const float* __restrict__ x, const float* __restrict__ qkv_w,
                const float* __restrict__ q_bias, const float* __restrict__ v_bias,
                __bf16* __restrict__ qb, __bf16* __restrict__ kb, __bf16* __restrict__ vtb) {
  int blk = blockIdx.x;            // 1024 blocks x 64 rows
  int row0 = blk * 64;
  int win = row0 >> 9;
  int nloc0 = row0 & 511;
  int tid = threadIdx.x, lane = tid & 31, wave = tid >> 5;
  __shared__ int rowoff[64];
  __shared__ float biasLds[288];   // concat(q_bias, 0, v_bias) staged once
  if (tid < 64) {
    int n = nloc0 + tid;
    int b = win >> 6, widx = win & 63;
    int zb = widx >> 4, hb = (widx >> 2) & 3, wb = widx & 3;
    int ez = n >> 6, eh = (n >> 3) & 7, ew = n & 7;
    int z = ((zb * 8 + ez) + 4) & 31;
    int h = ((hb * 8 + eh) + 4) & 31;
    int w = ((wb * 8 + ew) + 4) & 31;
    rowoff[tid] = (((b * 32 + z) * 32 + h) * 32 + w) * CDIM;
  }
  for (int i = tid; i < 288; i += 256)
    biasLds[i] = (i < 96) ? q_bias[i] : (i >= 192 ? v_bias[i - 192] : 0.f);
  __syncthreads();

  int rt = wave >> 1;              // 4 row tiles of 16
  int ctb = (wave & 1) * 9;        // 18 col tiles -> 9 per wave group
  int col = lane & 15, hl = lane >> 4;

  const float* xr = x + rowoff[rt * 16 + col];
  bf16x16 afr[3];
#pragma unroll
  for (int kc = 0; kc < 3; ++kc) afr[kc] = a_from_f32(xr + kc * 32, hl);

  for (int c = ctb; c < ctb + 9; ++c) {
    int c0 = c * 16;
    floatx8 acc = {};
#pragma unroll
    for (int kc = 0; kc < 3; ++kc) {
      bf16x16 bf = b_from_f32(qkv_w + (size_t)(c0 + col) * CDIM + kc * 32 + 16 * hl);
      acc = wmma_bf16(afr[kc], bf, acc);
    }
    float bias = biasLds[c0 + col];
    // section is uniform per 16-wide tile (96 % 16 == 0): no div/mod, wave-uniform branch
    int sect = (c0 >= 192) ? 2 : ((c0 >= 96) ? 1 : 0);
    int cc = c0 - sect * 96 + col;
    int head = cc >> 5, d = cc & 31;
    size_t base = (size_t)(win * 3 + head);
#pragma unroll
    for (int r = 0; r < 8; ++r) {
      int n = nloc0 + rt * 16 + r + 8 * hl;
      __bf16 bv = (__bf16)(acc[r] + bias);
      if (sect == 0)      qb[(base * 512 + n) * 32 + d] = bv;
      else if (sect == 1) kb[(base * 512 + n) * 32 + d] = bv;
      else                vtb[(base * 32 + d) * 512 + n] = bv;  // V stored transposed
    }
  }
}

// ---------- Kernel 3: cosine-attention row normalize (scale folded into q) ----------
__global__ void norm_qk_kernel(__bf16* __restrict__ qb, __bf16* __restrict__ kb,
                               const float* __restrict__ logit_scale) {
  int idx = blockIdx.x * blockDim.x + threadIdx.x;
  const int total = NWIN * 3 * NELEM;
  if (idx >= 2 * total) return;
  bool isK = idx >= total;
  int r = isK ? idx - total : idx;
  int head = (r >> 9) % 3;
  __bf16* p = (isK ? kb : qb) + (size_t)r * 32;
  float v[32], ss = 0.f;
#pragma unroll
  for (int d = 0; d < 32; ++d) { v[d] = (float)p[d]; ss += v[d] * v[d]; }
  float inv = 1.0f / fmaxf(sqrtf(ss), 1e-12f);
  if (!isK) inv *= __expf(fminf(logit_scale[head], 4.60517018599f));  // ln(100)
#pragma unroll
  for (int d = 0; d < 32; ++d) p[d] = (__bf16)(v[d] * inv);
}

// ---------- Kernel 4: window attention (scores live in LDS only) ----------
__global__ __launch_bounds__(256)
void attn_kernel(const __bf16* __restrict__ qb, const __bf16* __restrict__ kb,
                 const __bf16* __restrict__ vtb, const float* __restrict__ tbl,
                 float* __restrict__ attnout) {
  int bid = blockIdx.x;            // win*48 + head*16 + rb
  int rb = bid & 15;
  int head = (bid >> 4) % 3;
  int win = bid / 48;
  int tid = threadIdx.x, lane = tid & 31, wave = tid >> 5;

  __shared__ float S[32][512];          // 64 KB score tile
  __shared__ float sig[TBLN];           // 13.5 KB CPB bias (this head)
  __shared__ unsigned char regid[512];  // shift-mask region ids
  __shared__ float red[32][8];
  __shared__ float rowmax[32], rowsum[32];

  for (int t = tid; t < TBLN; t += 256)
    sig[t] = 16.0f / (1.0f + __expf(-tbl[t * 3 + head]));
  {
    int widx = win & 63;
    int zb = widx >> 4, hb = (widx >> 2) & 3, wb = widx & 3;
    for (int e = tid; e < 512; e += 256) {
      int z = zb * 8 + (e >> 6), h = hb * 8 + ((e >> 3) & 7), w = wb * 8 + (e & 7);
      int rz = (z < 24) ? 0 : (z < 28 ? 1 : 2);
      int rh = (h < 24) ? 0 : (h < 28 ? 1 : 2);
      int rw = (w < 24) ? 0 : (w < 28 ? 1 : 2);
      regid[e] = (unsigned char)(9 * rz + 3 * rh + rw);
    }
  }
  __syncthreads();

  size_t whb = (size_t)(win * 3 + head);
  const __bf16* qbase = qb + whb * 512 * 32;
  const __bf16* kbase = kb + whb * 512 * 32;
  const __bf16* vbase = vtb + whb * 32 * 512;
  int i0 = rb * 32;
  int col = lane & 15, hl = lane >> 4;

  // ---- S = qn @ knT + bias + mask : 2x32 tiles, 8 per wave ----
  {
    int rt = wave >> 2;
    int m0 = i0 + rt * 16;
    bf16x16 afr = a_from_bf16(qbase + (size_t)(m0 + col) * 32, hl);
    for (int ct = (wave & 3) * 8; ct < (wave & 3) * 8 + 8; ++ct) {
      int j0 = ct * 16;
      bf16x16 bfr = b_from_bf16(kbase + (size_t)(j0 + col) * 32 + 16 * hl);
      floatx8 acc = {};
      acc = wmma_bf16(afr, bfr, acc);
      int j = j0 + col;
      int jz = j >> 6, jh = (j >> 3) & 7, jw = j & 7;
      unsigned char rj = regid[j];
#pragma unroll
      for (int r = 0; r < 8; ++r) {
        int i = m0 + r + 8 * hl;
        int idx = ((i >> 6) - jz + 7) * 225 + (((i >> 3) & 7) - jh + 7) * 15 + ((i & 7) - jw + 7);
        float msk = (regid[i] != rj) ? -100.0f : 0.0f;
        S[i - i0][j] = acc[r] + sig[idx] + msk;
      }
    }
  }
  __syncthreads();

  // ---- softmax over 512, 8 threads per row ----
  {
    int sr = tid >> 3, sg = tid & 7;
    float mx = -1e30f;
    for (int j = sg * 64; j < sg * 64 + 64; ++j) mx = fmaxf(mx, S[sr][j]);
    red[sr][sg] = mx;
    __syncthreads();
    if (sg == 0) {
      float m = red[sr][0];
#pragma unroll
      for (int t = 1; t < 8; ++t) m = fmaxf(m, red[sr][t]);
      rowmax[sr] = m;
    }
    __syncthreads();
    float m = rowmax[sr], sm = 0.f;
    for (int j = sg * 64; j < sg * 64 + 64; ++j) { float e = __expf(S[sr][j] - m); S[sr][j] = e; sm += e; }
    red[sr][sg] = sm;
    __syncthreads();
    if (sg == 0) {
      float s = 0.f;
#pragma unroll
      for (int t = 0; t < 8; ++t) s += red[sr][t];
      rowsum[sr] = 1.0f / s;
    }
    __syncthreads();
    float inv = rowsum[sr];
    for (int j = sg * 64; j < sg * 64 + 64; ++j) S[sr][j] *= inv;
  }
  __syncthreads();

  // ---- out = P @ V : 2x2 tiles, K=512 (16 WMMA steps) ----
  if (wave < 4) {
    int rt = wave >> 1, ct = wave & 1;
    int m0 = rt * 16, d0 = ct * 16;
    floatx8 acc = {};
    for (int kc = 0; kc < 16; ++kc) {
      bf16x16 a = a_from_f32(&S[m0 + col][kc * 32], hl);             // f32->bf16 on the fly
      bf16x16 b = b_from_bf16(vbase + (size_t)(d0 + col) * 512 + kc * 32 + 16 * hl);
      acc = wmma_bf16(a, b, acc);
    }
    int ch = head * 32 + d0 + col;
#pragma unroll
    for (int r = 0; r < 8; ++r) {
      int n = i0 + m0 + r + 8 * hl;
      attnout[((size_t)win * 512 + n) * CDIM + ch] = acc[r];
    }
  }
}

// ---------- Kernel 5: output proj (WMMA) + window-reverse/unshift scatter ----------
__global__ __launch_bounds__(256)
void proj_kernel(const float* __restrict__ attnout, const float* __restrict__ proj_w,
                 const float* __restrict__ proj_b, float* __restrict__ ybuf) {
  int blk = blockIdx.x;
  int row0 = blk * 64;
  int win = row0 >> 9;
  int tid = threadIdx.x, lane = tid & 31, wave = tid >> 5;
  __shared__ int outoff[64];
  if (tid < 64) {
    int n = (row0 & 511) + tid;
    int b = win >> 6, widx = win & 63;
    int zb = widx >> 4, hb = (widx >> 2) & 3, wb = widx & 3;
    int z = ((zb * 8 + (n >> 6)) + 4) & 31;
    int h = ((hb * 8 + ((n >> 3) & 7)) + 4) & 31;
    int w = ((wb * 8 + (n & 7)) + 4) & 31;
    outoff[tid] = (((b * 32 + z) * 32 + h) * 32 + w) * CDIM;
  }
  __syncthreads();

  int rt = wave >> 1, ctb = (wave & 1) * 3;
  int col = lane & 15, hl = lane >> 4;
  const float* ar = attnout + (size_t)(row0 + rt * 16 + col) * CDIM;
  bf16x16 afr[3];
#pragma unroll
  for (int kc = 0; kc < 3; ++kc) afr[kc] = a_from_f32(ar + kc * 32, hl);

  for (int c = ctb; c < ctb + 3; ++c) {
    int c0 = c * 16;
    floatx8 acc = {};
#pragma unroll
    for (int kc = 0; kc < 3; ++kc) {
      bf16x16 bf = b_from_f32(proj_w + (size_t)(c0 + col) * CDIM + kc * 32 + 16 * hl);
      acc = wmma_bf16(afr[kc], bf, acc);
    }
    int cout = c0 + col;
    float bias = proj_b[cout];
#pragma unroll
    for (int r = 0; r < 8; ++r)
      ybuf[(size_t)outoff[rt * 16 + r + 8 * hl] + cout] = acc[r] + bias;
  }
}

// ---------- Kernels 6/8: post-norm residual out = base + LN(y), wave-per-row ----------
__global__ __launch_bounds__(256)
void addln_kernel(const float* __restrict__ base, const float* __restrict__ y,
                  const float* __restrict__ w, const float* __restrict__ b,
                  float* __restrict__ out, int rows) {
  int wave = threadIdx.x >> 5, lane = threadIdx.x & 31;
  int r = blockIdx.x * 8 + wave;
  if (r >= rows) return;
  const float* yr = y + (size_t)r * CDIM;
  const float* br = base + (size_t)r * CDIM;
  float v0 = yr[lane], v1 = yr[lane + 32], v2 = yr[lane + 64];
  float mu = wave_sum(v0 + v1 + v2) * (1.0f / CDIM);
  float d0 = v0 - mu, d1 = v1 - mu, d2 = v2 - mu;
  float var = wave_sum(d0 * d0 + d1 * d1 + d2 * d2) * (1.0f / CDIM);
  float inv = rsqrtf(var + LN_EPS);
  float* o = out + (size_t)r * CDIM;
  o[lane]      = br[lane]      + d0 * inv * w[lane]      + b[lane];
  o[lane + 32] = br[lane + 32] + d1 * inv * w[lane + 32] + b[lane + 32];
  o[lane + 64] = br[lane + 64] + d2 * inv * w[lane + 64] + b[lane + 64];
}

// ---------- Kernel 7: fused MLP fc1 -> leakyReLU -> fc2 (WMMA, hidden in LDS) ----------
__global__ __launch_bounds__(256)
void mlp_kernel(const float* __restrict__ x1, const float* __restrict__ fc1_w,
                const float* __restrict__ fc1_b, const float* __restrict__ fc2_w,
                const float* __restrict__ fc2_b, float* __restrict__ hbuf) {
  int blk = blockIdx.x;
  int row0 = blk * 64;
  int tid = threadIdx.x, lane = tid & 31, wave = tid >> 5;
  __shared__ __bf16 hid[64][HIDDEN];   // 48 KB
  int rt = wave >> 1;
  int col = lane & 15, hl = lane >> 4;

  // fc1: 64x384, K=96
  const float* xr = x1 + (size_t)(row0 + rt * 16 + col) * CDIM;
  bf16x16 afr[3];
#pragma unroll
  for (int kc = 0; kc < 3; ++kc) afr[kc] = a_from_f32(xr + kc * 32, hl);
  for (int c = (wave & 1) * 12; c < (wave & 1) * 12 + 12; ++c) {
    int c0 = c * 16;
    floatx8 acc = {};
#pragma unroll
    for (int kc = 0; kc < 3; ++kc) {
      bf16x16 bf = b_from_f32(fc1_w + (size_t)(c0 + col) * CDIM + kc * 32 + 16 * hl);
      acc = wmma_bf16(afr[kc], bf, acc);
    }
    int cout = c0 + col;
    float bias = fc1_b[cout];
#pragma unroll
    for (int r = 0; r < 8; ++r) {
      float v = acc[r] + bias;
      v = (v > 0.f) ? v : 0.01f * v;   // leaky relu
      hid[rt * 16 + r + 8 * hl][cout] = (__bf16)v;
    }
  }
  __syncthreads();

  // fc2: 64x96, K=384 (12 WMMA steps)
  for (int c = (wave & 1) * 3; c < (wave & 1) * 3 + 3; ++c) {
    int c0 = c * 16;
    floatx8 acc = {};
#pragma unroll
    for (int kc = 0; kc < 12; ++kc) {
      bf16x16 a = a_from_bf16(&hid[rt * 16 + col][kc * 32], hl);
      bf16x16 bf = b_from_f32(fc2_w + (size_t)(c0 + col) * HIDDEN + kc * 32 + 16 * hl);
      acc = wmma_bf16(a, bf, acc);
    }
    int cout = c0 + col;
    float bias = fc2_b[cout];
#pragma unroll
    for (int r = 0; r < 8; ++r)
      hbuf[(size_t)(row0 + rt * 16 + r + 8 * hl) * CDIM + cout] = acc[r] + bias;
  }
}

// ---------------------------------------------------------------------------
extern "C" void kernel_launch(void* const* d_in, const int* in_sizes, int n_in,
                              void* d_out, int out_size, void* d_ws, size_t ws_size,
                              hipStream_t stream) {
  (void)in_sizes; (void)n_in; (void)out_size; (void)ws_size;
  const float* x           = (const float*)d_in[0];
  const float* qkv_w       = (const float*)d_in[1];
  const float* q_bias      = (const float*)d_in[2];
  const float* v_bias      = (const float*)d_in[3];
  const float* logit_scale = (const float*)d_in[4];
  const float* cpb_w1      = (const float*)d_in[5];
  const float* cpb_b1      = (const float*)d_in[6];
  const float* cpb_w2      = (const float*)d_in[7];
  const float* proj_w      = (const float*)d_in[8];
  const float* proj_b      = (const float*)d_in[9];
  const float* norm1_w     = (const float*)d_in[10];
  const float* norm1_b     = (const float*)d_in[11];
  const float* fc1_w       = (const float*)d_in[12];
  const float* fc1_b       = (const float*)d_in[13];
  const float* fc2_w       = (const float*)d_in[14];
  const float* fc2_b       = (const float*)d_in[15];
  const float* norm2_w     = (const float*)d_in[16];
  const float* norm2_b     = (const float*)d_in[17];

  // workspace layout
  char* ws = (char*)d_ws;
  const size_t QKV_ELEMS = (size_t)NWIN * 3 * NELEM * HDIM;  // 6,291,456 per tensor
  float*  tbl  = (float*)ws;                                  // 40.5 KB (pad to 64 KB)
  __bf16* qb   = (__bf16*)(ws + (64 << 10));
  __bf16* kb   = qb + QKV_ELEMS;
  __bf16* vtb  = kb + QKV_ELEMS;
  float* attnout = (float*)(ws + (64 << 10) + 3 * QKV_ELEMS * sizeof(__bf16));
  float* ybuf = attnout + (size_t)ROWS * CDIM;
  float* x1   = ybuf + (size_t)ROWS * CDIM;
  float* hbuf = attnout;  // attnout dead after proj; reuse for MLP output

  cpb_kernel<<<(TBLN + 255) / 256, 256, 0, stream>>>(cpb_w1, cpb_b1, cpb_w2, tbl);
  qkv_kernel<<<ROWS / 64, 256, 0, stream>>>(x, qkv_w, q_bias, v_bias, qb, kb, vtb);
  {
    int rows = 2 * NWIN * 3 * NELEM;
    norm_qk_kernel<<<(rows + 255) / 256, 256, 0, stream>>>(qb, kb, logit_scale);
  }
  attn_kernel<<<NWIN * 3 * 16, 256, 0, stream>>>(qb, kb, vtb, tbl, attnout);
  proj_kernel<<<ROWS / 64, 256, 0, stream>>>(attnout, proj_w, proj_b, ybuf);
  addln_kernel<<<ROWS / 8, 256, 0, stream>>>(x, ybuf, norm1_w, norm1_b, x1, ROWS);
  mlp_kernel<<<ROWS / 64, 256, 0, stream>>>(x1, fc1_w, fc1_b, fc2_w, fc2_b, hbuf);
  addln_kernel<<<ROWS / 8, 256, 0, stream>>>(x1, hbuf, norm2_w, norm2_b, (float*)d_out, ROWS);
}
